// Decoder_63256278336156
// MI455X (gfx1250) — matrix operand order repaired and checked by reading, not compiled, not run on hardware
//
#include <hip/hip_runtime.h>
#include <hip/hip_bf16.h>

// ---------------- Problem constants (from reference) ----------------
#define Bq   128
#define Tm1  64
#define Mm   512          // encoder hidden
#define Pp   512          // decoder hidden
#define ROWS (Bq * Tm1)   // 8192

typedef __attribute__((ext_vector_type(16))) __bf16 bf16x16;
typedef __attribute__((ext_vector_type(8)))  __bf16 bf16x8;
typedef __attribute__((ext_vector_type(4)))  __bf16 bf16x4;
typedef __attribute__((ext_vector_type(8)))  float  floatx8;

union Frag { bf16x16 v; bf16x8 h[2]; };

// ---------------- fast activations (1 trans op each) ----------------
__device__ __forceinline__ float tanh_fast(float x) {
  return 1.0f - 2.0f / (__expf(2.0f * x) + 1.0f);   // tanh(x) = 1 - 2/(e^{2x}+1)
}
__device__ __forceinline__ float sigmoid_fast(float x) {
  return 1.0f / (1.0f + __expf(-x));
}

__device__ __forceinline__ floatx8 wmma_bf16(bf16x16 a, bf16x16 b, floatx8 c) {
  return __builtin_amdgcn_wmma_f32_16x16x32_bf16(false, a, false, b, (short)0, c, false, false);
}

// ---------------- f32 -> bf16 bulk convert (hoisted out of GEMM hot loop) ----------------
__global__ void cvt_f32_bf16(const float* __restrict__ src, __bf16* __restrict__ dst, int n4) {
  int i = (int)(blockIdx.x * blockDim.x + threadIdx.x);   // one float4 per thread
  if (i >= n4) return;
  float4 v = ((const float4*)src)[i];
  bf16x4 o;
  o[0] = (__bf16)v.x; o[1] = (__bf16)v.y; o[2] = (__bf16)v.z; o[3] = (__bf16)v.w;
  ((bf16x4*)dst)[i] = o;
}

// ---------------- WMMA bf16 GEMM, 32x32 super-tile per wave ----------------
// C[i,j] = sum_k A[i,k]*W[j,k] (+bias[j]); A,W bf16 row-major; M,N mult of 32; K mult of 32.
__global__ void gemm_wmma_bf16(const __bf16* __restrict__ A, int lda,
                               const __bf16* __restrict__ W, int ldw,
                               float* __restrict__ C, int ldc,
                               int Mrows, int Ncols, int K,
                               const float* __restrict__ bias) {
  int wave = (int)((blockIdx.x * blockDim.x + threadIdx.x) >> 5);
  int lane = (int)(threadIdx.x & 31);
  int ntn  = Ncols >> 5;
  int total = (Mrows >> 5) * ntn;
  if (wave >= total) return;                     // wave-uniform: EXEC stays all-1s

  int tm   = (wave / ntn) << 5;
  int tn   = (wave % ntn) << 5;
  int half = lane >> 4;
  int l16  = lane & 15;

  const __bf16* __restrict__ a0row = A + (size_t)(tm + l16) * lda;        // A rows for M-tile 0
  const __bf16* __restrict__ a1row = A + (size_t)(tm + 16 + l16) * lda;   // A rows for M-tile 1
  const __bf16* __restrict__ w0row = W + (size_t)(tn + l16) * ldw;        // B cols for N-tile 0
  const __bf16* __restrict__ w1row = W + (size_t)(tn + 16 + l16) * ldw;   // B cols for N-tile 1

  floatx8 acc00, acc01, acc10, acc11;
  float bi0 = bias ? bias[tn + l16] : 0.0f;
  float bi1 = bias ? bias[tn + 16 + l16] : 0.0f;
#pragma unroll
  for (int r = 0; r < 8; ++r) { acc00[r] = bi0; acc01[r] = bi1; acc10[r] = bi0; acc11[r] = bi1; }

  for (int k0 = 0; k0 < K; k0 += 32) {
    // A fragment: e[0..7] -> K = k0+8*half+j ; e[8..15] -> K = k0+16+8*half+j   (16B-aligned loads)
    Frag fa0, fa1, fb0, fb1;
    fa0.h[0] = *(const bf16x8*)(a0row + k0 + 8 * half);
    fa0.h[1] = *(const bf16x8*)(a0row + k0 + 16 + 8 * half);
    fa1.h[0] = *(const bf16x8*)(a1row + k0 + 8 * half);
    fa1.h[1] = *(const bf16x8*)(a1row + k0 + 16 + 8 * half);
    // B fragment: e[0..7] -> K = k0+8*half+j ; e[8..15] -> K = k0+16+8*half+j   (16B-aligned loads)
    fb0.h[0] = *(const bf16x8*)(w0row + k0 + 8 * half);
    fb0.h[1] = *(const bf16x8*)(w0row + k0 + 16 + 8 * half);
    fb1.h[0] = *(const bf16x8*)(w1row + k0 + 8 * half);
    fb1.h[1] = *(const bf16x8*)(w1row + k0 + 16 + 8 * half);
    if (k0 + 32 < K) {                       // warm L0 for next iteration
      __builtin_prefetch(a0row + k0 + 32, 0, 1);
      __builtin_prefetch(w0row + k0 + 32, 0, 1);
    }
    // four independent accumulation chains hide WMMA->WMMA RAW latency
    acc00 = wmma_bf16(fa0.v, fb0.v, acc00);
    acc01 = wmma_bf16(fa0.v, fb1.v, acc01);
    acc10 = wmma_bf16(fa1.v, fb0.v, acc10);
    acc11 = wmma_bf16(fa1.v, fb1.v, acc11);
  }

  // C/D layout: VGPR r -> row = 8*half + r, col = l16
  float* c00 = C + (size_t)tm * ldc + tn + l16;
#pragma unroll
  for (int r = 0; r < 8; ++r) {
    size_t ro0 = (size_t)(8 * half + r) * ldc;
    c00[ro0]            = acc00[r];
    c00[ro0 + 16]       = acc01[r];
    size_t ro1 = (size_t)(16 + 8 * half + r) * ldc;
    c00[ro1]            = acc10[r];
    c00[ro1 + 16]       = acc11[r];
  }
}

// ---------------- zero init of LSTM state (f32 + bf16 mirror) ----------------
__global__ void zero_dc(float* __restrict__ dc, __bf16* __restrict__ dcb, int n) {
  int i = (int)(blockIdx.x * blockDim.x + threadIdx.x);
  if (i < n) { dc[i] = 0.0f; dcb[i] = (__bf16)0.0f; }
}

// ---------------- attention score: e[b*64+t] = b_a2 + sum_m tanh(XE[b,t,m]+DC[b,m]) * w_a2[m] ----------------
__global__ void attn_score(const float* __restrict__ xe,      // [8192,512] (b_a1 folded in)
                           const float* __restrict__ dcp,     // [128,512]
                           const float* __restrict__ w_a2,    // [512]
                           const float* __restrict__ b_a2,    // [1]
                           float* __restrict__ e) {           // [8192]
  int wave = (int)((blockIdx.x * blockDim.x + threadIdx.x) >> 5);
  int lane = (int)(threadIdx.x & 31);
  if (wave >= ROWS) return;
  int b = wave >> 6;
  const float* __restrict__ xrow = xe  + (size_t)wave * Mm;
  const float* __restrict__ drow = dcp + (size_t)b * Mm;
  float acc = 0.0f;
#pragma unroll
  for (int i = 0; i < 4; ++i) {
    int m = i * 128 + lane * 4;
    float4 x = *(const float4*)(xrow + m);
    float4 d = *(const float4*)(drow + m);
    float4 w = *(const float4*)(w_a2 + m);
    acc += tanh_fast(x.x + d.x) * w.x + tanh_fast(x.y + d.y) * w.y
         + tanh_fast(x.z + d.z) * w.z + tanh_fast(x.w + d.w) * w.w;
  }
#pragma unroll
  for (int off = 16; off > 0; off >>= 1) acc += __shfl_xor(acc, off, 32);
  if (lane == 0) e[wave] = acc + b_a2[0];
}

// ---------------- softmax over flat chunks of 128 (faithful to the buggy (Tm1,B) reshape) ----------------
__global__ void softmax128(const float* __restrict__ e, float* __restrict__ beta) {
  __shared__ float s[128];
  int c = (int)blockIdx.x;        // 64 chunks
  int i = (int)threadIdx.x;       // 128 threads
  float v = e[c * 128 + i];
  s[i] = v; __syncthreads();
  for (int off = 64; off > 0; off >>= 1) { if (i < off) s[i] = fmaxf(s[i], s[i + off]); __syncthreads(); }
  float mx = s[0]; __syncthreads();
  float ex = __expf(v - mx);
  s[i] = ex; __syncthreads();
  for (int off = 64; off > 0; off >>= 1) { if (i < off) s[i] += s[i + off]; __syncthreads(); }
  beta[c * 128 + i] = ex / s[0];
}

// ---------------- context[b,:] = sum_t beta[b,t]*X[b,t,:]; y_tilde[b] = <context,W_fc[:M]> + y_prev[b,t]*W_fc[M] + b_fc ----------------
__global__ void context_ytilde(const float* __restrict__ beta,   // [8192] flat, beta[b*64+t]
                               const float* __restrict__ X,      // [128,64,512]
                               const float* __restrict__ W_fc,   // [513]
                               const float* __restrict__ b_fc,   // [1]
                               const float* __restrict__ y_prev, // [128,64]
                               int t,
                               float* __restrict__ context,      // [128,512]
                               float* __restrict__ y_tilde) {    // [128]
  int b = (int)blockIdx.x, tid = (int)threadIdx.x;  // 128 threads, one float4 each
  const float4* __restrict__ Xb = (const float4*)(X + (size_t)b * Tm1 * Mm);
  const float* __restrict__ bb = beta + (size_t)b * Tm1;
  float4 a = make_float4(0.f, 0.f, 0.f, 0.f);
  for (int tt = 0; tt < Tm1; ++tt) {
    float w = bb[tt];
    float4 x = Xb[tt * (Mm / 4) + tid];
    a.x += w * x.x; a.y += w * x.y; a.z += w * x.z; a.w += w * x.w;
  }
  ((float4*)(context + (size_t)b * Mm))[tid] = a;
  float4 wf = ((const float4*)W_fc)[tid];
  __shared__ float s[128];
  s[tid] = a.x * wf.x + a.y * wf.y + a.z * wf.z + a.w * wf.w;
  __syncthreads();
  for (int off = 64; off > 0; off >>= 1) { if (tid < off) s[tid] += s[tid + off]; __syncthreads(); }
  if (tid == 0) y_tilde[b] = s[0] + y_prev[(size_t)b * Tm1 + t] * W_fc[Mm] + b_fc[0];
}

// ---------------- LSTM pointwise update; writes f32 state + bf16 mirror for next-step GEMMs ----------------
__global__ void lstm_update(const float* __restrict__ gates,   // [128,2048] = d @ W_hh^T
                            const float* __restrict__ b_ih,    // [2048]
                            const float* __restrict__ b_hh,    // [2048]
                            const float* __restrict__ W_ih,    // [2048] (input size 1)
                            const float* __restrict__ y_tilde, // [128]
                            float* __restrict__ dc,            // [128,1024] = [d|c]
                            __bf16* __restrict__ dcb) {        // bf16 mirror
  int idx = (int)(blockIdx.x * blockDim.x + threadIdx.x);      // 65536
  int b = idx >> 9, p = idx & 511;
  float yt = y_tilde[b];
  const float* __restrict__ g = gates + (size_t)b * 4 * Pp;
  float gi = g[p]          + b_ih[p]          + b_hh[p]          + yt * W_ih[p];
  float gf = g[Pp + p]     + b_ih[Pp + p]     + b_hh[Pp + p]     + yt * W_ih[Pp + p];
  float gg = g[2 * Pp + p] + b_ih[2 * Pp + p] + b_hh[2 * Pp + p] + yt * W_ih[2 * Pp + p];
  float go = g[3 * Pp + p] + b_ih[3 * Pp + p] + b_hh[3 * Pp + p] + yt * W_ih[3 * Pp + p];
  float c  = dc[(size_t)b * 1024 + Pp + p];
  float cn = sigmoid_fast(gf) * c + sigmoid_fast(gi) * tanh_fast(gg);
  float dn = sigmoid_fast(go) * tanh_fast(cn);
  dc[(size_t)b * 1024 + p]       = dn;
  dc[(size_t)b * 1024 + Pp + p]  = cn;
  dcb[(size_t)b * 1024 + p]      = (__bf16)dn;
  dcb[(size_t)b * 1024 + Pp + p] = (__bf16)cn;
}

// ---------------- final: out[b] = <[d,context], W_fin> + b_fin ----------------
__global__ void final_out(const float* __restrict__ dc,      // [128,1024]
                          const float* __restrict__ context, // [128,512]
                          const float* __restrict__ W_fin,   // [1024]
                          const float* __restrict__ b_fin,   // [1]
                          float* __restrict__ out) {         // [128]
  int b = (int)blockIdx.x, tid = (int)threadIdx.x;  // 256 threads
  const float* __restrict__ d   = dc + (size_t)b * 1024;
  const float* __restrict__ ctx = context + (size_t)b * Mm;
  float part = d[tid]         * W_fin[tid]
             + d[256 + tid]   * W_fin[256 + tid]
             + ctx[tid]       * W_fin[Pp + tid]
             + ctx[256 + tid] * W_fin[Pp + 256 + tid];
  __shared__ float s[256];
  s[tid] = part; __syncthreads();
  for (int off = 128; off > 0; off >>= 1) { if (tid < off) s[tid] += s[tid + off]; __syncthreads(); }
  if (tid == 0) out[b] = s[0] + b_fin[0];
}

// ---------------- host orchestration ----------------
extern "C" void kernel_launch(void* const* d_in, const int* in_sizes, int n_in,
                              void* d_out, int out_size, void* d_ws, size_t ws_size,
                              hipStream_t stream) {
  (void)in_sizes; (void)n_in; (void)out_size; (void)ws_size;
  const float* X_enc  = (const float*)d_in[0];   // (128,64,512)
  const float* y_prev = (const float*)d_in[1];   // (128,64)
  const float* W_a1   = (const float*)d_in[2];   // (512,1536)
  const float* b_a1   = (const float*)d_in[3];   // (512)
  const float* W_a2   = (const float*)d_in[4];   // (1,512)
  const float* b_a2   = (const float*)d_in[5];   // (1)
  const float* W_ih   = (const float*)d_in[6];   // (2048,1)
  const float* W_hh   = (const float*)d_in[7];   // (2048,512)
  const float* b_ih   = (const float*)d_in[8];   // (2048)
  const float* b_hh   = (const float*)d_in[9];   // (2048)
  const float* W_fc   = (const float*)d_in[10];  // (1,513)
  const float* b_fc   = (const float*)d_in[11];  // (1)
  const float* W_fin  = (const float*)d_in[12];  // (1,1024)
  const float* b_fin  = (const float*)d_in[13];  // (1)
  float* out = (float*)d_out;                    // (128,1) f32

  // workspace layout (bytes)
  char* ws = (char*)d_ws;
  float*  XE_proj  = (float*) (ws);                  // 8192*512 f32      = 16,777,216
  __bf16* X_bf     = (__bf16*)(ws + 16777216);       // 8192*512 bf16     =  8,388,608
  __bf16* Wa1_bf   = (__bf16*)(ws + 25165824);       // 512*1536 bf16     =  1,572,864
  __bf16* Whh_bf   = (__bf16*)(ws + 26738688);       // 2048*512 bf16     =  2,097,152
  float*  dc       = (float*) (ws + 28835840);       // 128*1024 f32      =    524,288
  __bf16* dc_bf    = (__bf16*)(ws + 29360128);       // 128*1024 bf16     =    262,144
  float*  dc_proj  = (float*) (ws + 29622272);       // 128*512 f32       =    262,144
  float*  e_flat   = (float*) (ws + 29884416);       // 8192 f32
  float*  beta     = (float*) (ws + 29917184);       // 8192 f32
  float*  context  = (float*) (ws + 29949952);       // 128*512 f32       =    262,144
  float*  y_tilde  = (float*) (ws + 30212096);       // 128 f32
  float*  gates    = (float*) (ws + 30212608);       // 128*2048 f32      =  1,048,576

  // one-time bf16 conversions (hoisted out of all GEMM hot loops)
  cvt_f32_bf16<<<(ROWS * Mm / 4 + 255) / 256, 256, 0, stream>>>(X_enc, X_bf, ROWS * Mm / 4);
  cvt_f32_bf16<<<(Mm * 1536 / 4 + 255) / 256, 256, 0, stream>>>(W_a1, Wa1_bf, Mm * 1536 / 4);
  cvt_f32_bf16<<<(2048 * Pp / 4 + 255) / 256, 256, 0, stream>>>(W_hh, Whh_bf, 2048 * Pp / 4);

  // d = c = 0 (every call: deterministic)
  zero_dc<<<(Bq * 1024 + 255) / 256, 256, 0, stream>>>(dc, dc_bf, Bq * 1024);

  // Hoisted: XE_proj[b,t,m] = X_enc[b,t,:] . W_a1[m, 2P:2P+M] + b_a1[m]
  // 32x32 super-tiles: (8192/32)*(512/32) = 4096 waves -> 512 blocks of 8 waves
  gemm_wmma_bf16<<<512, 256, 0, stream>>>(X_bf, Mm, Wa1_bf + 2 * Pp, 2 * Pp + Mm,
                                          XE_proj, Mm, ROWS, Mm, Mm, b_a1);

  for (int t = 0; t < Tm1; ++t) {
    // dc_proj[b,m] = [d|c] . W_a1[m, 0:2P]   (4*16 = 64 waves -> 8 blocks)
    gemm_wmma_bf16<<<8, 256, 0, stream>>>(dc_bf, 1024, Wa1_bf, 2 * Pp + Mm,
                                          dc_proj, Mm, Bq, Mm, 2 * Pp, nullptr);
    // scores (8192 waves -> 1024 blocks)
    attn_score<<<1024, 256, 0, stream>>>(XE_proj, dc_proj, W_a2, b_a2, e_flat);
    // buggy-reshape softmax: 64 chunks of 128 flat elements
    softmax128<<<64, 128, 0, stream>>>(e_flat, beta);
    // context + y_tilde
    context_ytilde<<<Bq, 128, 0, stream>>>(beta, X_enc, W_fc, b_fc, y_prev, t,
                                           context, y_tilde);
    // gates = d @ W_hh^T  (A = dc_bf with lda=1024, K=512 covers d half; 4*64 waves -> 32 blocks)
    gemm_wmma_bf16<<<32, 256, 0, stream>>>(dc_bf, 1024, Whh_bf, Pp,
                                           gates, 4 * Pp, Bq, 4 * Pp, Pp, nullptr);
    // LSTM pointwise
    lstm_update<<<(Bq * Pp + 255) / 256, 256, 0, stream>>>(gates, b_ih, b_hh, W_ih,
                                                           y_tilde, dc, dc_bf);
  }

  final_out<<<Bq, 256, 0, stream>>>(dc, context, W_fin, b_fin, out);
}
